// ResidualVectorQuantizer_68977174773795
// MI455X (gfx1250) — matrix operand order repaired and checked by reading, not compiled, not run on hardware
//
#include <hip/hip_runtime.h>

// ---------------------------------------------------------------------------
// Residual VQ for MI455X (gfx1250, wave32, WMMA).
//   x:         (B=16, S=2048, D=64) f32
//   codebooks: (K=8, M=2048, D=64)  f32
//   out:       quantized (B,S,D) f32 || indices (K,B,S) i32  (concatenated)
//
// Per stage: fused GEMM+argmin+update kernel.
//  - Block stages the full stage codebook into LDS as f16 (WMMA-ready, rows
//    padded to 72 halfwords to dodge bank conflicts) plus a c2[m] table.
//  - Each wave owns 32 rows (two 16-row A tiles, f16 in registers), sweeps
//    M in 16-column tiles with 4x v_wmma_f32_16x16x32_f16 per tile.
//  - B tile + c2 are double-buffered in registers (software pipeline) so DS
//    latency overlaps the previous tile's WMMA + argmin VALU.
//  - d2 = c2[m] - 2*cross (||r||^2 is row-constant -> drops from argmin).
// ---------------------------------------------------------------------------

typedef __attribute__((ext_vector_type(16))) _Float16 v16h;
typedef __attribute__((ext_vector_type(8)))  float    v8f;

#define RVQ_B 16
#define RVQ_S 2048
#define RVQ_D 64
#define RVQ_K 8
#define RVQ_M 2048
#define RVQ_R (RVQ_B * RVQ_S)          // 32768 rows

#define ROWTILES        2                 // 16-row A tiles per wave
#define ROWS_PER_WAVE   (16 * ROWTILES)   // 32
#define WAVES_PER_BLOCK 8
#define ROWS_PER_BLOCK  (ROWS_PER_WAVE * WAVES_PER_BLOCK)  // 256

#define CB_STRIDE_H 72                    // f16 elems per LDS codebook row (64+8 pad)

static __device__ inline v16h cvt16(const float4 a, const float4 b,
                                    const float4 c, const float4 d) {
  v16h r;
  r[0]  = (_Float16)a.x; r[1]  = (_Float16)a.y; r[2]  = (_Float16)a.z; r[3]  = (_Float16)a.w;
  r[4]  = (_Float16)b.x; r[5]  = (_Float16)b.y; r[6]  = (_Float16)b.z; r[7]  = (_Float16)b.w;
  r[8]  = (_Float16)c.x; r[9]  = (_Float16)c.y; r[10] = (_Float16)c.z; r[11] = (_Float16)c.w;
  r[12] = (_Float16)d.x; r[13] = (_Float16)d.y; r[14] = (_Float16)d.z; r[15] = (_Float16)d.w;
  return r;
}

__global__ __launch_bounds__(256) void rvq_step_kernel(
    const float* __restrict__ rin,    // (R, D) residual in (== x on stage 0)
    const float* __restrict__ cb,     // (M, D) f32 codebook for this stage
    float* __restrict__ rout,         // (R, D) residual out (may alias rin)
    int* __restrict__ idx_out)        // (R)
{
  // ~297 KB static LDS (<= 320 KB/workgroup on CDNA5): one block per WGP.
  __shared__ _Float16 s_cbh[RVQ_M * CB_STRIDE_H];   // f16 codebook, padded rows
  __shared__ float    s_c2[RVQ_M];                  // ||c||^2 per codeword
  __shared__ int      s_idx[ROWS_PER_BLOCK];

  const int tid  = threadIdx.x;
  const int wave = tid >> 5;
  const int lane = tid & 31;
  const int hf   = lane >> 4;        // 0 = lanes 0-15, 1 = lanes 16-31
  const int l16  = lane & 15;
  const int rowBase = blockIdx.x * ROWS_PER_BLOCK + wave * ROWS_PER_WAVE;

  // ==== Stage codebook into LDS: f32 -> f16 + c2, once per block ====
#pragma unroll
  for (int i = 0; i < RVQ_M / ROWS_PER_BLOCK; ++i) {      // 8 rows per thread
    const int m = tid + i * ROWS_PER_BLOCK;
    const float4* src = (const float4*)(cb + (size_t)m * RVQ_D);
    float a0 = 0.f, a1 = 0.f, a2 = 0.f, a3 = 0.f;         // 4-way ILP for c2
    union { uint4 q[8]; _Float16 h[64]; } buf;
#pragma unroll
    for (int j = 0; j < 16; ++j) {
      const float4 v = src[j];
      a0 = fmaf(v.x, v.x, a0); a1 = fmaf(v.y, v.y, a1);
      a2 = fmaf(v.z, v.z, a2); a3 = fmaf(v.w, v.w, a3);
      buf.h[4 * j + 0] = (_Float16)v.x; buf.h[4 * j + 1] = (_Float16)v.y;
      buf.h[4 * j + 2] = (_Float16)v.z; buf.h[4 * j + 3] = (_Float16)v.w;
    }
    s_c2[m] = (a0 + a1) + (a2 + a3);
    uint4* dst = (uint4*)&s_cbh[m * CB_STRIDE_H];         // 16B-aligned (144B rows)
#pragma unroll
    for (int j = 0; j < 8; ++j) dst[j] = buf.q[j];
  }

  // ==== Load residual rows, convert once into WMMA A layout ====
  // A 16x32 (MxK), this code's contiguous k order (B matches): lane(<16)=row M, holds K = d0..d0+15,
  // d0 = 0 (lanes 0-15) / 16 (lanes 16-31), per K-chunk of 32.
  v16h a[ROWTILES][2];
  const int d0 = hf * 16;
#pragma unroll
  for (int t = 0; t < ROWTILES; ++t) {
    const float* rp = rin + (size_t)(rowBase + t * 16 + l16) * RVQ_D;
#pragma unroll
    for (int c = 0; c < 2; ++c) {
      const float* p = rp + c * 32 + d0;
      float4 x0 = *(const float4*)(p);
      float4 x1 = *(const float4*)(p + 4);
      float4 x2 = *(const float4*)(p + 8);
      float4 x3 = *(const float4*)(p + 12);
      a[t][c] = cvt16(x0, x1, x2, x3);
    }
  }

  float best[ROWTILES][8];
  int   bidx[ROWTILES][8];
#pragma unroll
  for (int t = 0; t < ROWTILES; ++t)
#pragma unroll
    for (int j = 0; j < 8; ++j) { best[t][j] = 3.4e38f; bidx[t][j] = 0; }

  __syncthreads();   // codebook staged

  // ==== Sweep codebook in 16-column tiles (double-buffered B + c2) ====
  // B 32x16 (KxN), contiguous k order as for A: lane(<16)=col N holds K=0..15; lanes 16-31 K=16..31.
  const int koff = hf * 16;
  union BU { uint4 q[2]; v16h v; };
  BU   pb0[2], pb1[2];
  float pc2[2];

  {  // prologue: prefetch tile 0 into buffer 0
    const _Float16* bp = &s_cbh[l16 * CB_STRIDE_H + koff];
    pb0[0].q[0] = *(const uint4*)(bp);
    pb0[0].q[1] = *(const uint4*)(bp + 8);
    pb1[0].q[0] = *(const uint4*)(bp + 32);
    pb1[0].q[1] = *(const uint4*)(bp + 40);
    pc2[0] = s_c2[l16];
  }

  int col = l16;
#pragma unroll 2
  for (int mt = 0; mt < RVQ_M; mt += 16, col += 16) {
    const int cur = (mt >> 4) & 1;             // constant after unroll-by-2
    const int nxt = cur ^ 1;

    // prefetch next tile (wraps to col0 on last iter; harmless, branch-free)
    const int ncol = (col + 16) & (RVQ_M - 1);
    const _Float16* bp = &s_cbh[ncol * CB_STRIDE_H + koff];
    pb0[nxt].q[0] = *(const uint4*)(bp);        // d = koff .. koff+7
    pb0[nxt].q[1] = *(const uint4*)(bp + 8);    // d = koff+8 .. koff+15
    pb1[nxt].q[0] = *(const uint4*)(bp + 32);   // d = 32+koff ..
    pb1[nxt].q[1] = *(const uint4*)(bp + 40);
    pc2[nxt] = s_c2[ncol];

    v8f acc[ROWTILES];
#pragma unroll
    for (int t = 0; t < ROWTILES; ++t) {
      acc[t] = (v8f){};
      acc[t] = __builtin_amdgcn_wmma_f32_16x16x32_f16(
          false, a[t][0], false, pb0[cur].v, (short)0, acc[t], false, false);
      acc[t] = __builtin_amdgcn_wmma_f32_16x16x32_f16(
          false, a[t][1], false, pb1[cur].v, (short)0, acc[t], false, false);
    }

    // d2 = c2 - 2*cross ; running argmin (col strictly increases per lane,
    // so '<' keeps the first occurrence within a lane).
    const float c2 = pc2[cur];
#pragma unroll
    for (int t = 0; t < ROWTILES; ++t) {
#pragma unroll
      for (int j = 0; j < 8; ++j) {
        const float v = fmaf(-2.0f, acc[t][j], c2);
        if (v < best[t][j]) { best[t][j] = v; bidx[t][j] = col; }
      }
    }
  }

  // ==== Butterfly argmin across each 16-lane half (C/D layout: lanes 0-15
  //      hold rows M=j, lanes 16-31 hold rows M=8+j) ====
#pragma unroll
  for (int t = 0; t < ROWTILES; ++t) {
#pragma unroll
    for (int j = 0; j < 8; ++j) {
      float v = best[t][j];
      int   i = bidx[t][j];
#pragma unroll
      for (int m = 0; m < 4; ++m) {
        const int mask = 1 << m;             // 1,2,4,8 : stays within half
        const float ov = __shfl_xor(v, mask);
        const int   oi = __shfl_xor(i, mask);
        if (ov < v || (ov == v && oi < i)) { v = ov; i = oi; }
      }
      bidx[t][j] = i;
    }
  }

  if (l16 == 0) {                             // lanes 0 and 16 publish results
#pragma unroll
    for (int t = 0; t < ROWTILES; ++t)
#pragma unroll
      for (int j = 0; j < 8; ++j)
        s_idx[wave * ROWS_PER_WAVE + t * 16 + hf * 8 + j] = bidx[t][j];
  }
  __syncthreads();

  // ==== Gather winning codeword (f32, exact), update residual, emit index ====
  const int row = rowBase + lane;             // one row per lane (32 rows/wave)
  const int ci  = s_idx[wave * ROWS_PER_WAVE + lane];
  idx_out[row] = ci;

  const float4* cq = (const float4*)(cb + (size_t)ci * RVQ_D);
  const float4* rq = (const float4*)(rin + (size_t)row * RVQ_D);
  float4*       oq = (float4*)(rout + (size_t)row * RVQ_D);
#pragma unroll
  for (int i = 0; i < RVQ_D / 4; ++i) {
    const float4 rv = rq[i];
    const float4 cv = cq[i];
    oq[i] = make_float4(rv.x - cv.x, rv.y - cv.y, rv.z - cv.z, rv.w - cv.w);
  }
}

// quantized = x - residual (residual currently stored in q; overwrite in place)
__global__ __launch_bounds__(256) void rvq_finalize_kernel(
    const float* __restrict__ x, float* __restrict__ q, int n)
{
  const int i = blockIdx.x * blockDim.x + threadIdx.x;
  if (i < n) q[i] = x[i] - q[i];
}

extern "C" void kernel_launch(void* const* d_in, const int* in_sizes, int n_in,
                              void* d_out, int out_size, void* d_ws, size_t ws_size,
                              hipStream_t stream) {
  (void)in_sizes; (void)n_in; (void)out_size; (void)d_ws; (void)ws_size;

  const float* x   = (const float*)d_in[0];              // (B,S,D)
  const float* cbs = (const float*)d_in[1];              // (K,M,D)
  float* qout = (float*)d_out;                           // (B,S,D) quantized
  int*   idx  = (int*)((float*)d_out + (size_t)RVQ_R * RVQ_D);  // (K,B,S)

  const dim3 block(ROWS_PER_BLOCK);
  const dim3 grid(RVQ_R / ROWS_PER_BLOCK);               // 128 blocks

  for (int k = 0; k < RVQ_K; ++k) {
    const float* rin = (k == 0) ? x : qout;              // in-place residual
    const float* cb  = cbs + (size_t)k * RVQ_M * RVQ_D;
    rvq_step_kernel<<<grid, block, 0, stream>>>(rin, cb, qout, idx + (size_t)k * RVQ_R);
  }

  const int n = RVQ_R * RVQ_D;
  rvq_finalize_kernel<<<dim3(n / 256), dim3(256), 0, stream>>>(x, qout, n);
}
